// CausalMultiheadsSelfAttentionWithRope_21277267984729
// MI455X (gfx1250) — matrix operand order repaired and checked by reading, not compiled, not run on hardware
//
#include <hip/hip_runtime.h>
#include <hip/hip_bf16.h>

// Problem constants (match reference)
#define D_MODEL   1024
#define NUM_HEADS 16
#define SEQ       2048
#define BATCH     4
#define DKH       64          // head dim
#define MROWS     (BATCH*SEQ) // 8192

typedef __attribute__((ext_vector_type(16))) __bf16 v16bf;
typedef __attribute__((ext_vector_type(8)))  float  v8f;

union FragAB { v16bf v; uint4 q[2]; };

// ---- CDNA5 async global->LDS staging (ASYNCcnt), guarded for toolchain ----
#if defined(__has_builtin)
#if __has_builtin(__builtin_amdgcn_global_load_async_to_lds_b128) && \
    __has_builtin(__builtin_amdgcn_s_wait_asynccnt)
#define USE_ASYNC_LDS 1
#endif
#endif
#ifndef USE_ASYNC_LDS
#define USE_ASYNC_LDS 0
#endif

#if USE_ASYNC_LDS
// Builtin signature (from hipcc diagnostic): first param is
// 'int __attribute__((vector_size(16))) __device__ *' i.e. v4i in addrspace(1).
typedef int v4i_t __attribute__((vector_size(16)));
typedef __attribute__((address_space(1))) v4i_t gv4i_t;
typedef __attribute__((address_space(3))) v4i_t lv4i_t;
__device__ __forceinline__ void async_copy_b128(const unsigned short* g,
                                                unsigned short* l) {
  __builtin_amdgcn_global_load_async_to_lds_b128((gv4i_t*)g, (lv4i_t*)l, 0, 0);
}
__device__ __forceinline__ void async_wait0() {
  __builtin_amdgcn_s_wait_asynccnt(0);
}
#endif

__device__ __forceinline__ v8f wmma_bf16f32(v16bf a, v16bf b, v8f c) {
  // D = A(16x32 bf16) * B(32x16 bf16) + C(16x16 f32)
  return __builtin_amdgcn_wmma_f32_16x16x32_bf16(false, a, false, b, (short)0, c,
                                                 false, false);
}

__device__ __forceinline__ unsigned short f32_to_bf16(float f) {
  unsigned int u = __float_as_uint(f);
  u += 0x7FFFu + ((u >> 16) & 1u); // round-to-nearest-even
  return (unsigned short)(u >> 16);
}
__device__ __forceinline__ float bf16_to_f32(unsigned short h) {
  return __uint_as_float(((unsigned int)h) << 16);
}

// ---- DPP16 row reductions (16-lane rows), pure VALU: no LDS round trips ----
template <int CTRL>
__device__ __forceinline__ float fdpp(float x) {
  int xi = __float_as_int(x);
  return __int_as_float(__builtin_amdgcn_update_dpp(xi, xi, CTRL, 0xf, 0xf, true));
}
__device__ __forceinline__ float rowmax16(float v) {
  v = fmaxf(v, fdpp<0x121>(v)); // row_ror:1
  v = fmaxf(v, fdpp<0x122>(v)); // row_ror:2
  v = fmaxf(v, fdpp<0x124>(v)); // row_ror:4
  v = fmaxf(v, fdpp<0x128>(v)); // row_ror:8
  return v;
}
__device__ __forceinline__ float rowsum16(float v) {
  v += fdpp<0x121>(v);
  v += fdpp<0x122>(v);
  v += fdpp<0x124>(v);
  v += fdpp<0x128>(v);
  return v;
}

// ---------------------------------------------------------------------------
// f32 -> bf16 conversion
// ---------------------------------------------------------------------------
__global__ __launch_bounds__(256) void cvt_f32_bf16(const float* __restrict__ src,
                                                    unsigned short* __restrict__ dst,
                                                    int n) {
  int i = blockIdx.x * 256 + threadIdx.x;
  if (i < n) dst[i] = f32_to_bf16(src[i]);
}

// ---------------------------------------------------------------------------
// GEMM: C[M,N] = A[M,K] * W[N,K]^T   (A,W bf16 row-major; C f32 or bf16)
// Block 128x64, 256 threads (8 waves), wave tile 32x32 (2x2 WMMA), K-step 32.
// Async path: double-buffered LDS, prefetch tile i+1 while computing tile i.
// ---------------------------------------------------------------------------
template <bool OUT_BF16>
__global__ __launch_bounds__(256) void gemm_bf16(const unsigned short* __restrict__ A,
                                                 const unsigned short* __restrict__ W,
                                                 void* __restrict__ Cout,
                                                 int Mdim, int Ndim, int Kdim) {
#if USE_ASYNC_LDS
  __shared__ unsigned short sA[2][128 * 40]; // row stride 40 elems (80B, 16B-aligned)
  __shared__ unsigned short sB[2][64 * 40];
#else
  __shared__ unsigned short sA1[128 * 40];
  __shared__ unsigned short sB1[64 * 40];
#endif

  const int m0 = blockIdx.x * 128;
  const int n0 = blockIdx.y * 64;
  const int t = threadIdx.x;
  const int wv = t >> 5, lane = t & 31;
  const int wm = wv >> 1, wn = wv & 1; // 4x2 wave grid
  const int hlf = lane >> 4, l16 = lane & 15;

  v8f acc[2][2];
#pragma unroll
  for (int i = 0; i < 2; i++)
#pragma unroll
    for (int j = 0; j < 2; j++)
#pragma unroll
      for (int r = 0; r < 8; r++) acc[i][j][r] = 0.0f;

  auto compute_tile = [&](const unsigned short* Abuf, const unsigned short* Bbuf) {
    FragAB af[2], bfr[2];
#pragma unroll
    for (int i = 0; i < 2; i++) {
      const unsigned short* p = &Abuf[(wm * 32 + i * 16 + l16) * 40 + hlf * 8];
      af[i].q[0] = *(const uint4*)p;        // K = hlf*8 .. +7
      af[i].q[1] = *(const uint4*)(p + 16); // K = 16+hlf*8 .. +7
    }
#pragma unroll
    for (int j = 0; j < 2; j++) {
      const unsigned short* p = &Bbuf[(wn * 32 + j * 16 + l16) * 40 + hlf * 8];
      bfr[j].q[0] = *(const uint4*)p;       // K = hlf*8 .. +7
      bfr[j].q[1] = *(const uint4*)(p + 16); // K = 16+hlf*8 .. +7
    }
#pragma unroll
    for (int i = 0; i < 2; i++)
#pragma unroll
      for (int j = 0; j < 2; j++)
        acc[i][j] = wmma_bf16f32(af[i].v, bfr[j].v, acc[i][j]);
  };

#if USE_ASYNC_LDS
  auto stage_async = [&](int kb, int buf) {
#pragma unroll
    for (int it = 0; it < 2; it++) {
      int i = t + it * 256;
      int row = i >> 2, seg = i & 3;
      async_copy_b128(A + (size_t)(m0 + row) * Kdim + kb + seg * 8,
                      &sA[buf][row * 40 + seg * 8]);
    }
    {
      int row = t >> 2, seg = t & 3;
      async_copy_b128(W + (size_t)(n0 + row) * Kdim + kb + seg * 8,
                      &sB[buf][row * 40 + seg * 8]);
    }
  };
  const int nk = Kdim >> 5;
  stage_async(0, 0);
  async_wait0();
  __syncthreads();
  for (int i = 0; i < nk; i++) {
    if (i + 1 < nk) stage_async((i + 1) << 5, (i + 1) & 1); // prefetch next tile
    compute_tile(sA[i & 1], sB[i & 1]);                     // overlap with WMMAs
    async_wait0();
    __syncthreads();
  }
#else
  for (int kb = 0; kb < Kdim; kb += 32) {
#pragma unroll
    for (int it = 0; it < 2; it++) {
      int i = t + it * 256;
      int row = i >> 2, seg = i & 3;
      uint4 d = *(const uint4*)(A + (size_t)(m0 + row) * Kdim + kb + seg * 8);
      *(uint4*)(&sA1[row * 40 + seg * 8]) = d;
    }
    {
      int row = t >> 2, seg = t & 3;
      uint4 d = *(const uint4*)(W + (size_t)(n0 + row) * Kdim + kb + seg * 8);
      *(uint4*)(&sB1[row * 40 + seg * 8]) = d;
    }
    __syncthreads();
    compute_tile(sA1, sB1);
    __syncthreads();
  }
#endif

  // Store C: lane l -> n = l16, rows r + 8*hlf
#pragma unroll
  for (int i = 0; i < 2; i++)
#pragma unroll
    for (int j = 0; j < 2; j++) {
      int n = n0 + wn * 32 + j * 16 + l16;
#pragma unroll
      for (int r = 0; r < 8; r++) {
        int m = m0 + wm * 32 + i * 16 + hlf * 8 + r;
        float val = acc[i][j][r];
        if constexpr (OUT_BF16)
          ((unsigned short*)Cout)[(size_t)m * Ndim + n] = f32_to_bf16(val);
        else
          ((float*)Cout)[(size_t)m * Ndim + n] = val;
      }
    }
}

// ---------------------------------------------------------------------------
// RoPE + rearrange [b,s,h,d] -> [b,h,s,d]; Q scaled by 1/sqrt(dk); V copied.
// ---------------------------------------------------------------------------
__global__ __launch_bounds__(256) void rope_rearrange(
    const unsigned short* __restrict__ Qraw, const unsigned short* __restrict__ Kraw,
    const unsigned short* __restrict__ Vraw, const int* __restrict__ pos,
    unsigned short* __restrict__ Qr, unsigned short* __restrict__ Kr,
    unsigned short* __restrict__ Vr) {
  int idx = blockIdx.x * 256 + threadIdx.x;
  if (idx >= BATCH * NUM_HEADS * SEQ * 32) return;
  int i = idx & 31;
  int t1 = idx >> 5;
  int s = t1 & (SEQ - 1);
  int t2 = t1 >> 11;
  int h = t2 & (NUM_HEADS - 1);
  int b = t2 >> 4;

  float p = (float)pos[s];
  // theta^(-i/32) = exp(-i * ln(10000)/32)
  float ang = p * __expf(-(float)i * 0.2878231366f);
  float sn, cs;
  __sincosf(ang, &sn, &cs);

  size_t srcb = (size_t)(b * SEQ + s) * D_MODEL + h * DKH;
  size_t dstb = ((size_t)(b * NUM_HEADS + h) * SEQ + s) * DKH;

  float qe = bf16_to_f32(Qraw[srcb + 2 * i]);
  float qo = bf16_to_f32(Qraw[srcb + 2 * i + 1]);
  Qr[dstb + 2 * i]     = f32_to_bf16((qe * cs - qo * sn) * 0.125f);
  Qr[dstb + 2 * i + 1] = f32_to_bf16((qe * sn + qo * cs) * 0.125f);

  float ke = bf16_to_f32(Kraw[srcb + 2 * i]);
  float ko = bf16_to_f32(Kraw[srcb + 2 * i + 1]);
  Kr[dstb + 2 * i]     = f32_to_bf16(ke * cs - ko * sn);
  Kr[dstb + 2 * i + 1] = f32_to_bf16(ke * sn + ko * cs);

  Vr[dstb + 2 * i]     = Vraw[srcb + 2 * i];
  Vr[dstb + 2 * i + 1] = Vraw[srcb + 2 * i + 1];
}

// ---------------------------------------------------------------------------
// Flash attention: block = (128 q rows) x (b,h). 8 waves, 16 q-rows each.
// KV streamed in tiles of 32 with online softmax. All matmuls via WMMA bf16.
// ---------------------------------------------------------------------------
__global__ __launch_bounds__(256) void attn_flash(
    const unsigned short* __restrict__ Qr, const unsigned short* __restrict__ Kr,
    const unsigned short* __restrict__ Vr, unsigned short* __restrict__ Out) {
  __shared__ unsigned short sK[32 * 72];    // [kv][d], stride 72 (144B)
  __shared__ unsigned short sVt[64 * 40];   // [d][kv], stride 40 (80B)
  __shared__ unsigned short sP[8][16 * 40]; // per-wave P 16x32, stride 40

  const int qt = blockIdx.x;
  const int bh = blockIdx.y;
  const int b = bh >> 4, h = bh & 15;
  const int t = threadIdx.x, wv = t >> 5, lane = t & 31;
  const int hlf = lane >> 4, l16 = lane & 15;
  const int q0 = qt * 128;
  const int qw = q0 + wv * 16;

  const size_t headoff = (size_t)bh * SEQ * DKH;

  // Preload Q fragments (A-layout straight from global, rows contiguous)
  FragAB qf[2];
#pragma unroll
  for (int f = 0; f < 2; f++) {
    const unsigned short* p = Qr + headoff + (size_t)(qw + l16) * DKH + f * 32 + hlf * 8;
    qf[f].q[0] = *(const uint4*)p;
    qf[f].q[1] = *(const uint4*)(p + 16);
  }

  v8f o[4];
  float Mrow[8], Lrow[8];
#pragma unroll
  for (int c = 0; c < 4; c++)
#pragma unroll
    for (int r = 0; r < 8; r++) o[c][r] = 0.0f;
#pragma unroll
  for (int r = 0; r < 8; r++) { Mrow[r] = -1e30f; Lrow[r] = 0.0f; }

  const int nkb = (q0 >> 5) + 4; // cover kv <= q0+127
  for (int kb = 0; kb < nkb; kb++) {
    const int kv0 = kb * 32;
    // Stage K [32][64] (async if available) and V transposed [64][32]
    {
      int row = t >> 3, seg = t & 7;
      const unsigned short* gK = Kr + headoff + (size_t)(kv0 + row) * DKH + seg * 8;
#if USE_ASYNC_LDS
      async_copy_b128(gK, &sK[row * 72 + seg * 8]);
#else
      uint4 dk4 = *(const uint4*)gK;
      *(uint4*)(&sK[row * 72 + seg * 8]) = dk4;
#endif
      uint4 dv4 = *(const uint4*)(Vr + headoff + (size_t)(kv0 + row) * DKH + seg * 8);
      const unsigned short* pv = (const unsigned short*)&dv4;
#pragma unroll
      for (int e = 0; e < 8; e++) sVt[(seg * 8 + e) * 40 + row] = pv[e];
    }
#if USE_ASYNC_LDS
    async_wait0();
#endif
    __syncthreads();

    // Waves whose q rows are all below this kv block contribute nothing.
    if (kv0 <= qw + 15) {
      v8f zf;
#pragma unroll
      for (int r = 0; r < 8; r++) zf[r] = 0.0f;
      v8f sc[2];
      const bool needMask = (kv0 + 31) > qw;
#pragma unroll
      for (int c = 0; c < 2; c++) {
        FragAB kf[2];
#pragma unroll
        for (int f = 0; f < 2; f++) {
          const unsigned short* p = &sK[(c * 16 + l16) * 72 + f * 32 + hlf * 8];
          kf[f].q[0] = *(const uint4*)p;
          kf[f].q[1] = *(const uint4*)(p + 16);
        }
        sc[c] = wmma_bf16f32(qf[0].v, kf[0].v, zf);
        sc[c] = wmma_bf16f32(qf[1].v, kf[1].v, sc[c]);
        if (needMask) {
          int kvpos = kv0 + c * 16 + l16;
#pragma unroll
          for (int r = 0; r < 8; r++) {
            int qpos = qw + hlf * 8 + r;
            if (kvpos > qpos) sc[c][r] = -1e30f;
          }
        }
      }

      // Online softmax (DPP row reductions across the 16 lanes of each half)
      float alpha[8];
#pragma unroll
      for (int r = 0; r < 8; r++) {
        float m = rowmax16(fmaxf(sc[0][r], sc[1][r]));
        float Mnew = fmaxf(Mrow[r], m);
        alpha[r] = __expf(Mrow[r] - Mnew);
        Mrow[r] = Mnew;
      }
#pragma unroll
      for (int c = 0; c < 2; c++)
#pragma unroll
        for (int r = 0; r < 8; r++) sc[c][r] = __expf(sc[c][r] - Mrow[r]);
#pragma unroll
      for (int r = 0; r < 8; r++) {
        float s = rowsum16(sc[0][r] + sc[1][r]);
        Lrow[r] = Lrow[r] * alpha[r] + s;
      }
#pragma unroll
      for (int c = 0; c < 4; c++)
#pragma unroll
        for (int r = 0; r < 8; r++) o[c][r] *= alpha[r];

      // P (C-layout) -> per-wave LDS -> A-fragment layout
#pragma unroll
      for (int c = 0; c < 2; c++)
#pragma unroll
        for (int r = 0; r < 8; r++)
          sP[wv][(hlf * 8 + r) * 40 + c * 16 + l16] = f32_to_bf16(sc[c][r]);
      FragAB pf;
      {
        const unsigned short* p = &sP[wv][l16 * 40 + hlf * 8];
        pf.q[0] = *(const uint4*)p;
        pf.q[1] = *(const uint4*)(p + 16);
      }

      // O += P(16x32) * V(32x64), over 4 dk chunks of 16
#pragma unroll
      for (int c2 = 0; c2 < 4; c2++) {
        FragAB vf;
        const unsigned short* p = &sVt[(c2 * 16 + l16) * 40 + hlf * 8];
        vf.q[0] = *(const uint4*)p;
        vf.q[1] = *(const uint4*)(p + 16);
        o[c2] = wmma_bf16f32(pf.v, vf.v, o[c2]);
      }
    }
    __syncthreads();
  }

  // Normalize and store to [b, s, h*64+d] as bf16 for the final GEMM
#pragma unroll
  for (int c2 = 0; c2 < 4; c2++)
#pragma unroll
    for (int r = 0; r < 8; r++) {
      int srow = qw + hlf * 8 + r;
      float val = o[c2][r] / Lrow[r];
      Out[(size_t)(b * SEQ + srow) * D_MODEL + h * DKH + c2 * 16 + l16] =
          f32_to_bf16(val);
    }
}

// ---------------------------------------------------------------------------
// Host-side orchestration
// ---------------------------------------------------------------------------
extern "C" void kernel_launch(void* const* d_in, const int* in_sizes, int n_in,
                              void* d_out, int out_size, void* d_ws, size_t ws_size,
                              hipStream_t stream) {
  (void)in_sizes; (void)n_in; (void)out_size; (void)ws_size;
  const float* Wq = (const float*)d_in[0];
  const float* Wk = (const float*)d_in[1];
  const float* Wv = (const float*)d_in[2];
  const float* Wo = (const float*)d_in[3];
  const float* X  = (const float*)d_in[4];
  const int* pos  = (const int*)d_in[5];

  char* ws = (char*)d_ws;
  // Workspace map (bytes), total 120 MiB
  unsigned short* Xb   = (unsigned short*)(ws + 0);                      // 16 MiB
  unsigned short* Wqb  = (unsigned short*)(ws + (size_t)16 * (1 << 20)); // 2 MiB each
  unsigned short* Wkb  = (unsigned short*)(ws + (size_t)18 * (1 << 20));
  unsigned short* Wvb  = (unsigned short*)(ws + (size_t)20 * (1 << 20));
  unsigned short* Wob  = (unsigned short*)(ws + (size_t)22 * (1 << 20));
  unsigned short* Qraw = (unsigned short*)(ws + (size_t)24 * (1 << 20)); // 16 MiB (reused as attn-out)
  unsigned short* Kraw = (unsigned short*)(ws + (size_t)40 * (1 << 20));
  unsigned short* Vraw = (unsigned short*)(ws + (size_t)56 * (1 << 20));
  unsigned short* Qrr  = (unsigned short*)(ws + (size_t)72 * (1 << 20));
  unsigned short* Krr  = (unsigned short*)(ws + (size_t)88 * (1 << 20));
  unsigned short* Vrr  = (unsigned short*)(ws + (size_t)104 * (1 << 20));

  // 1) f32 -> bf16
  cvt_f32_bf16<<<(MROWS * D_MODEL + 255) / 256, 256, 0, stream>>>(X, Xb, MROWS * D_MODEL);
  cvt_f32_bf16<<<(D_MODEL * D_MODEL + 255) / 256, 256, 0, stream>>>(Wq, Wqb, D_MODEL * D_MODEL);
  cvt_f32_bf16<<<(D_MODEL * D_MODEL + 255) / 256, 256, 0, stream>>>(Wk, Wkb, D_MODEL * D_MODEL);
  cvt_f32_bf16<<<(D_MODEL * D_MODEL + 255) / 256, 256, 0, stream>>>(Wv, Wvb, D_MODEL * D_MODEL);
  cvt_f32_bf16<<<(D_MODEL * D_MODEL + 255) / 256, 256, 0, stream>>>(Wo, Wob, D_MODEL * D_MODEL);

  // 2) QKV projections (WMMA GEMM, bf16 out)
  dim3 gg(MROWS / 128, D_MODEL / 64);
  gemm_bf16<true><<<gg, 256, 0, stream>>>(Xb, Wqb, Qraw, MROWS, D_MODEL, D_MODEL);
  gemm_bf16<true><<<gg, 256, 0, stream>>>(Xb, Wkb, Kraw, MROWS, D_MODEL, D_MODEL);
  gemm_bf16<true><<<gg, 256, 0, stream>>>(Xb, Wvb, Vraw, MROWS, D_MODEL, D_MODEL);

  // 3) RoPE + [b,s,h,d] -> [b,h,s,d] (Q pre-scaled by 1/sqrt(64))
  rope_rearrange<<<(BATCH * NUM_HEADS * SEQ * 32) / 256, 256, 0, stream>>>(
      Qraw, Kraw, Vraw, pos, Qrr, Krr, Vrr);

  // 4) Flash attention (WMMA) -> writes bf16 into Qraw (reused as attn-out)
  dim3 ga(SEQ / 128, BATCH * NUM_HEADS);
  attn_flash<<<ga, 256, 0, stream>>>(Qrr, Krr, Vrr, Qraw);

  // 5) Output projection -> f32 into d_out
  gemm_bf16<false><<<gg, 256, 0, stream>>>(Qraw, Wob, (float*)d_out, MROWS, D_MODEL,
                                           D_MODEL);
}
